// FlashMLP_59219009077886
// MI455X (gfx1250) — matrix operand order repaired and checked by reading, not compiled, not run on hardware
//
#include <hip/hip_runtime.h>

// ---------------------------------------------------------------------------
// FlashMLP for MI455X (gfx1250, wave32, WMMA).
// out[b,h,s,:] = (silu(Q·K^T * s) * (Q·U^T * s)) · V      (GQA: kh = h/4)
// bf16 WMMA with f32 accumulation, flash-style streaming over I (BI=32).
// Each of 8 waves owns TWO 16-row M-strips (BM=256) so every LDS B-fragment
// (K/U/V columns) feeds two WMMAs -> 48 WMMAs per I-tile per wave.
// ---------------------------------------------------------------------------

typedef __bf16        v16bf __attribute__((ext_vector_type(16)));
typedef float         v8f   __attribute__((ext_vector_type(8)));
typedef float         v4f   __attribute__((ext_vector_type(4)));
typedef unsigned int  v4u   __attribute__((ext_vector_type(4)));

union FragU { v16bf bf; unsigned short u[16]; unsigned int w[8]; v4u q[2]; };

#define B_   2
#define H_   16
#define S_   2048
#define D_   128
#define I_   2048
#define BM   256          // M rows per workgroup (8 waves x 2 strips x 16)
#define BI   32           // I-tile (K-dim of down GEMM)
#define NIT  (I_ / BI)
#define KSTR 136          // padded LDS row stride (bf16) for K/U tiles
#define VSTR 40           // padded LDS row stride (bf16) for V^T / A scratch
#define SCALE 0.08838834764831845f

// f32 pair -> packed bf16x2: round-half-up (+0x8000), merge hi16s with v_perm
__device__ __forceinline__ unsigned int pk2(float a, float b) {
  unsigned int ua = __builtin_bit_cast(unsigned int, a) + 0x8000u;
  unsigned int ub = __builtin_bit_cast(unsigned int, b) + 0x8000u;
  // result byte0,1 = ua bytes 2,3 (sel 0x02,0x03: src1); byte2,3 = ub bytes 2,3
  return __builtin_amdgcn_perm(ub, ua, 0x07060302u);
}
// single f32 -> bf16 (compiler folds the >>16 into ds_store_b16_d16_hi)
__device__ __forceinline__ unsigned short f2bf1(float f) {
  unsigned int x = __builtin_bit_cast(unsigned int, f) + 0x8000u;
  return (unsigned short)(x >> 16);
}

// 32 contiguous bytes from LDS -> one 16-element bf16 fragment
__device__ __forceinline__ v16bf ld32(const unsigned short* p) {
  FragU f;
  f.q[0] = *(const v4u*)(p);
  f.q[1] = *(const v4u*)(p + 8);
  return f.bf;
}

#define WMMA_BF16(A, Bm, C) \
  __builtin_amdgcn_wmma_f32_16x16x32_bf16(false, (A), false, (Bm), (short)0, (C), false, false)

// gate: A = silu(M*s) * (N*s), written to per-wave scratch in C-matrix layout
// silu via v_rcp_f32 (avoids the IEEE divide expansion; ~1ulp is plenty vs bf16)
__device__ __forceinline__ void gate_store(unsigned short* as, v8f m0, v8f m1,
                                           v8f n0, v8f n1, int mm, int hi) {
#pragma unroll
  for (int t = 0; t < 2; ++t) {
    v8f mv = t ? m1 : m0;
    v8f nv = t ? n1 : n0;
#pragma unroll
    for (int v = 0; v < 8; ++v) {
      float x  = mv[v] * SCALE;
      float y  = nv[v] * SCALE;
      float sg = x * __builtin_amdgcn_rcpf(1.0f + __expf(-x));   // silu
      as[(v + hi * 8) * VSTR + t * 16 + mm] = f2bf1(sg * y);
    }
  }
}

__global__ __launch_bounds__(256)
void flashmlp_wmma_kernel(const float* __restrict__ Q, const float* __restrict__ K,
                          const float* __restrict__ U, const float* __restrict__ V,
                          float* __restrict__ O) {
  __shared__ unsigned short Kt[BI * KSTR];          //  8704 B  K tile [i][d]
  __shared__ unsigned short Ut[BI * KSTR];          //  8704 B  U tile [i][d]
  __shared__ unsigned short Vt[D_ * VSTR];          // 10240 B  V tile transposed [d][i]
  __shared__ unsigned short As[8 * 2 * 16 * VSTR];  // 20480 B  per-wave/strip gate scratch

  const int tid  = threadIdx.x;
  const int lane = tid & 31;
  const int wv   = tid >> 5;        // wave 0..7
  const int mm   = lane & 15;       // row (A/C) or column (B) within a 16-tile
  const int hi   = lane >> 4;       // lane half selects K sub-chunk

  const int bid   = blockIdx.x;
  const int stile = bid & 7;            // S/BM = 8
  const int h     = (bid >> 3) & 15;
  const int b     = bid >> 7;
  const int kh    = h >> 2;             // GQA group (G = 4)

  const int s0 = stile * BM + wv * 16;  // strip0 rows; strip1 = s0 + 128

  // ---- Q fragments: 2 strips x 16 rows x 128 (4 K-chunks), pinned in VGPRs ----
  v16bf qf0[4], qf1[4];
#pragma unroll
  for (int sx = 0; sx < 2; ++sx) {
    const float* qrow =
        Q + (((size_t)(b * H_ + h) * S_ + (size_t)(s0 + sx * 128 + mm)) * D_);
#pragma unroll
    for (int c = 0; c < 4; ++c) {
      FragU f;
      const v4f* ga = (const v4f*)(qrow + c * 32 + hi * 16);       // K 0..7 / 16..23
      const v4f* gb = (const v4f*)(qrow + c * 32 + hi * 16 + 8);   // K 8..15 / 24..31
      v4f a0 = ga[0], a1 = ga[1], b0 = gb[0], b1 = gb[1];
      f.w[0] = pk2(a0.x, a0.y); f.w[1] = pk2(a0.z, a0.w);
      f.w[2] = pk2(a1.x, a1.y); f.w[3] = pk2(a1.z, a1.w);
      f.w[4] = pk2(b0.x, b0.y); f.w[5] = pk2(b0.z, b0.w);
      f.w[6] = pk2(b1.x, b1.y); f.w[7] = pk2(b1.z, b1.w);
      if (sx == 0) qf0[c] = f.bf; else qf1[c] = f.bf;
    }
  }

  const v8f vzero = {0.f, 0.f, 0.f, 0.f, 0.f, 0.f, 0.f, 0.f};
  v8f accO0[8], accO1[8];
#pragma unroll
  for (int j = 0; j < 8; ++j) { accO0[j] = vzero; accO1[j] = vzero; }

  unsigned short* as0 = &As[(wv * 2 + 0) * 16 * VSTR];
  unsigned short* as1 = &As[(wv * 2 + 1) * 16 * VSTR];

  // staging coordinates (per thread, loop-invariant)
  const int rr  = tid >> 3;            // K/U: i row 0..31
  const int ck  = (tid & 7) * 16;      // K/U: d col {0,16,..,112}
  const int ip  = (tid & 15) * 2;      // V: i pair base {0,2,..,30}
  const int cv  = (tid >> 4) * 8;      // V: d group {0,8,..,120}

#pragma unroll 1
  for (int it = 0; it < NIT; ++it) {
    const int i0 = it * BI;

    // ---- cooperative staging: f32 global -> bf16 LDS ----
    {
      const size_t gku = ((size_t)kh * I_ + (size_t)(i0 + rr)) * D_ + ck;
      const v4f* k4 = (const v4f*)(K + gku);
      const v4f* u4 = (const v4f*)(U + gku);
      FragU fk, fu;
      {
        v4f a0 = k4[0], a1 = k4[1], b0 = k4[2], b1 = k4[3];
        fk.w[0] = pk2(a0.x, a0.y); fk.w[1] = pk2(a0.z, a0.w);
        fk.w[2] = pk2(a1.x, a1.y); fk.w[3] = pk2(a1.z, a1.w);
        fk.w[4] = pk2(b0.x, b0.y); fk.w[5] = pk2(b0.z, b0.w);
        fk.w[6] = pk2(b1.x, b1.y); fk.w[7] = pk2(b1.z, b1.w);
      }
      {
        v4f a0 = u4[0], a1 = u4[1], b0 = u4[2], b1 = u4[3];
        fu.w[0] = pk2(a0.x, a0.y); fu.w[1] = pk2(a0.z, a0.w);
        fu.w[2] = pk2(a1.x, a1.y); fu.w[3] = pk2(a1.z, a1.w);
        fu.w[4] = pk2(b0.x, b0.y); fu.w[5] = pk2(b0.z, b0.w);
        fu.w[6] = pk2(b1.x, b1.y); fu.w[7] = pk2(b1.z, b1.w);
      }
      *(v4u*)&Kt[rr * KSTR + ck]     = fk.q[0];
      *(v4u*)&Kt[rr * KSTR + ck + 8] = fk.q[1];
      *(v4u*)&Ut[rr * KSTR + ck]     = fu.q[0];
      *(v4u*)&Ut[rr * KSTR + ck + 8] = fu.q[1];

      // V transposed: read two adjacent i rows, pack (i,i+1) pairs -> b32 stores
      const float* vr0 = V + ((size_t)kh * I_ + (size_t)(i0 + ip)) * D_ + cv;
      const float* vr1 = vr0 + D_;
      v4f x0 = ((const v4f*)vr0)[0], x1 = ((const v4f*)vr0)[1];
      v4f y0 = ((const v4f*)vr1)[0], y1 = ((const v4f*)vr1)[1];
#pragma unroll
      for (int j = 0; j < 4; ++j)
        *(unsigned int*)&Vt[(cv + j) * VSTR + ip] = pk2(x0[j], y0[j]);
#pragma unroll
      for (int j = 0; j < 4; ++j)
        *(unsigned int*)&Vt[(cv + 4 + j) * VSTR + ip] = pk2(x1[j], y1[j]);

      // prefetch next I-tile into the cache hierarchy (global_prefetch_b8)
      if (it + 1 < NIT) {
        __builtin_prefetch(K + gku + (size_t)BI * D_, 0, 3);
        __builtin_prefetch(U + gku + (size_t)BI * D_, 0, 3);
        __builtin_prefetch(vr0 + (size_t)BI * D_, 0, 3);
      }
    }
    __syncthreads();

    // ---- score GEMMs: each B-fragment feeds both M-strips ----
    v8f aM00 = vzero, aM01 = vzero, aN00 = vzero, aN01 = vzero;   // strip0
    v8f aM10 = vzero, aM11 = vzero, aN10 = vzero, aN11 = vzero;   // strip1
#pragma unroll
    for (int c = 0; c < 4; ++c) {
      const int koff = c * 32 + hi * 16;   // B-frag: 16 consecutive K per lane
      v16bf kb0 = ld32(&Kt[(mm)      * KSTR + koff]);
      v16bf kb1 = ld32(&Kt[(16 + mm) * KSTR + koff]);
      v16bf ub0 = ld32(&Ut[(mm)      * KSTR + koff]);
      v16bf ub1 = ld32(&Ut[(16 + mm) * KSTR + koff]);
      aM00 = WMMA_BF16(qf0[c], kb0, aM00);
      aM10 = WMMA_BF16(qf1[c], kb0, aM10);
      aM01 = WMMA_BF16(qf0[c], kb1, aM01);
      aM11 = WMMA_BF16(qf1[c], kb1, aM11);
      aN00 = WMMA_BF16(qf0[c], ub0, aN00);
      aN10 = WMMA_BF16(qf1[c], ub0, aN10);
      aN01 = WMMA_BF16(qf0[c], ub1, aN01);
      aN11 = WMMA_BF16(qf1[c], ub1, aN11);
    }

    // ---- gate both strips; C-layout -> A-layout via per-wave LDS scratch ----
    gate_store(as0, aM00, aM01, aN00, aN01, mm, hi);
    gate_store(as1, aM10, aM11, aN10, aN11, mm, hi);
    __asm__ volatile("" ::: "memory");  // LDS is in-order per wave; fence compiler

    FragU fa0, fa1;
    fa0.q[0] = *(const v4u*)&as0[mm * VSTR + hi * 16];
    fa0.q[1] = *(const v4u*)&as0[mm * VSTR + hi * 16 + 8];
    fa1.q[0] = *(const v4u*)&as1[mm * VSTR + hi * 16];
    fa1.q[1] = *(const v4u*)&as1[mm * VSTR + hi * 16 + 8];
    v16bf af0 = fa0.bf, af1 = fa1.bf;

    // ---- down GEMM: out += A · V ; each V-fragment feeds both strips ----
#pragma unroll
    for (int j = 0; j < 8; ++j) {
      v16bf vb = ld32(&Vt[(j * 16 + mm) * VSTR + hi * 16]);
      accO0[j] = WMMA_BF16(af0, vb, accO0[j]);
      accO1[j] = WMMA_BF16(af1, vb, accO1[j]);
    }
    __syncthreads();
  }

  // ---- writeback: 2 strips x 16x128 f32 per wave ----
  float* obase = O + ((size_t)(b * H_ + h) * S_) * D_;
#pragma unroll
  for (int j = 0; j < 8; ++j) {
#pragma unroll
    for (int v = 0; v < 8; ++v) {
      const int m = v + hi * 8;
      obase[(size_t)(s0 + m) * D_ + j * 16 + mm]       = accO0[j][v];
      obase[(size_t)(s0 + 128 + m) * D_ + j * 16 + mm] = accO1[j][v];
    }
  }
}

extern "C" void kernel_launch(void* const* d_in, const int* in_sizes, int n_in,
                              void* d_out, int out_size, void* d_ws, size_t ws_size,
                              hipStream_t stream) {
  (void)in_sizes; (void)n_in; (void)out_size; (void)d_ws; (void)ws_size;
  const float* Q = (const float*)d_in[0];
  const float* K = (const float*)d_in[1];
  const float* U = (const float*)d_in[2];
  const float* V = (const float*)d_in[3];
  float* O = (float*)d_out;
  dim3 grid(B_ * H_ * (S_ / BM));   // 2*16*8 = 256 workgroups
  flashmlp_wmma_kernel<<<grid, 256, 0, stream>>>(Q, K, U, V, O);
}
